// EDCM_31095563223633
// MI455X (gfx1250) — compile-verified, matrix operand order checked by reading, not run on hardware
//
#include <hip/hip_runtime.h>
#include <hip/hip_bf16.h>

typedef __attribute__((ext_vector_type(16))) __bf16 v16bf;
typedef __attribute__((ext_vector_type(8)))  __bf16 v8bf;
typedef __attribute__((ext_vector_type(8)))  float  v8f;

#define BN_EPS 1e-5f
constexpr int B  = 16;
constexpr int C1 = 256;
constexpr int C2 = 256;
constexpr int H  = 64;
constexpr int W  = 64;
constexpr int AC = 16;
constexpr int HP = 66;          // padded spatial dim (halo of 1)
constexpr int NPIX = H * W;
constexpr int TP = 258;         // packx LDS channel pitch (odd dword stride)

__device__ __forceinline__ float sigf(float z) { return 1.f / (1.f + __expf(-z)); }

// ---------------------------------------------------------------------------
// Kernel 1: global average pool. One block per (b,c) plane.
// ---------------------------------------------------------------------------
__global__ void pool_kernel(const float* __restrict__ x, float* __restrict__ pooled) {
    __shared__ float red[256];
    const int plane = blockIdx.x;                 // b*C1 + c
    const float* p = x + (size_t)plane * NPIX;
    float s = 0.f;
    for (int t = threadIdx.x; t < NPIX; t += 256) s += p[t];
    red[threadIdx.x] = s;
    __syncthreads();
    for (int k = 128; k > 0; k >>= 1) {
        if (threadIdx.x < k) red[threadIdx.x] += red[threadIdx.x + k];
        __syncthreads();
    }
    if (threadIdx.x == 0) pooled[plane] = red[0] * (1.f / NPIX);
}

// ---------------------------------------------------------------------------
// Kernel 2: attention trunk + heads + BN folding. Single block, 256 threads.
// ---------------------------------------------------------------------------
__global__ void attn_kernel(const float* __restrict__ pooled,
                            const float* fc_w, const float* g1, const float* b1,
                            const float* m1, const float* v1,
                            const float* ch_w, const float* ch_b,
                            const float* fil_w, const float* fil_b,
                            const float* sp_w, const float* sp_b,
                            const float* bias, const float* g2, const float* b2,
                            const float* m2, const float* v2,
                            float* chattn, float* alpha, float* beta, float* spv) {
    __shared__ float aL[B * AC];
    const int tid = threadIdx.x;
    {   // a = relu(bn1(pooled @ fc_w.T)) ; one (b,j) per thread (256 == B*AC)
        const int b = tid >> 4, j = tid & 15;
        const float* pw = fc_w + j * C1;
        const float* pp = pooled + b * C1;
        float s = 0.f;
        for (int c = 0; c < C1; c++) s += pp[c] * pw[c];
        const float sc = g1[j] * rsqrtf(v1[j] + BN_EPS);
        s = (s - m1[j]) * sc + b1[j];
        aL[tid] = fmaxf(s, 0.f);
    }
    __syncthreads();
    for (int idx = tid; idx < B * C1; idx += 256) {     // channel head
        const int b = idx >> 8, i = idx & 255;
        const float* w = ch_w + i * AC;
        const float* av = aL + b * AC;
        float s = ch_b[i];
        for (int j = 0; j < AC; j++) s += av[j] * w[j];
        chattn[idx] = sigf(s);
    }
    for (int idx = tid; idx < B * C2; idx += 256) {     // filter head * bn scale
        const int b = idx >> 8, o = idx & 255;
        const float* w = fil_w + o * AC;
        const float* av = aL + b * AC;
        float s = fil_b[o];
        for (int j = 0; j < AC; j++) s += av[j] * w[j];
        const float sc = g2[o] * rsqrtf(v2[o] + BN_EPS);
        alpha[idx] = sigf(s) * sc;
    }
    if (tid < C2) {                                     // folded bias/BN shift
        const float sc = g2[tid] * rsqrtf(v2[tid] + BN_EPS);
        beta[tid] = bias[tid] * sc + b2[tid] - m2[tid] * sc;
    }
    if (tid < B * 9) {                                  // spatial head (KN=1 -> ker==1)
        const int b = tid / 9, uv = tid % 9;
        const float* w = sp_w + uv * AC;
        const float* av = aL + b * AC;
        float s = sp_b[uv];
        for (int j = 0; j < AC; j++) s += av[j] * w[j];
        spv[tid] = sigf(s);
    }
}

// ---------------------------------------------------------------------------
// Kernel 3: NCHW f32 -> channel-last padded bf16, channel-attn folded.
// xs[b][py(66)][px(66)][ch(256)]; interior (py,px) = (y+1, x+1); halo = 0.
// One block per (b,py) row. Transpose via padded LDS tile (conflict-free).
// ---------------------------------------------------------------------------
__global__ void packx_kernel(const float* __restrict__ x,
                             const float* __restrict__ chattn,
                             __bf16* __restrict__ xs) {
    __shared__ __bf16 tile[HP * TP];              // [px][ch], pitch 258
    const int b  = blockIdx.x / HP;
    const int py = blockIdx.x % HP;
    const int tid = threadIdx.x;
    __bf16* orow = xs + ((size_t)(b * HP + py)) * HP * C1;

    if (py == 0 || py == HP - 1) {                // zero halo rows, b128 stores
        uint4 z = {0u, 0u, 0u, 0u};
        uint4* o4 = (uint4*)orow;
        for (int idx = tid; idx < HP * C1 * 2 / 16; idx += 256) o4[idx] = z;
        return;
    }
    const int y = py - 1;
    // zero halo columns px=0, px=65
    if (tid < 256) { tile[tid] = (__bf16)0.f; tile[65 * TP + tid] = (__bf16)0.f; }
    // phase 1: coalesced NCHW read, scaled, transposed store into LDS
    for (int idx = tid; idx < C1 * W; idx += 256) {
        const int ch = idx >> 6, xc = idx & 63;
        const float v = x[((size_t)(b * C1 + ch) * H + y) * W + xc] * chattn[b * C1 + ch];
        tile[(xc + 1) * TP + ch] = (__bf16)v;
    }
    __syncthreads();
    // phase 2: contiguous-channel read, b128 global store
    for (int idx = tid; idx < HP * 32; idx += 256) {
        const int px = idx >> 5, q = idx & 31;
        const unsigned* t32 = (const unsigned*)&tile[px * TP + q * 8];
        uint4 d;
        d.x = t32[0]; d.y = t32[1]; d.z = t32[2]; d.w = t32[3];
        ((uint4*)(orow + (size_t)px * C1))[q] = d;
    }
}

// ---------------------------------------------------------------------------
// Kernel 4: pre-pack W0 into per-lane WMMA A-fragment layout (bf16).
// apack[((uv*16 + mf)*8 + ko)*32 + lane][16]
// ---------------------------------------------------------------------------
__global__ void packw_kernel(const float* __restrict__ weight,
                             __bf16* __restrict__ apack) {
    const int blk = blockIdx.x;                   // (uv*16 + mf)*8 + ko
    const int uv = blk >> 7;
    const int ko = blk & 7;
    const int mfbase = ((blk >> 3) & 15) * 16;
    const int lane = threadIdx.x;                 // 32 threads = one wave
    const int m = mfbase + (lane & 15);
    const int kh = (lane >> 4) * 8;
    __bf16* out = apack + ((size_t)blk * 32 + lane) * 16;
    #pragma unroll
    for (int e = 0; e < 16; e++) {
        const int kk = (e < 8) ? (kh + e) : (16 + kh + (e - 8));
        const int i = ko * 32 + kk;
        out[e] = (__bf16)weight[((size_t)m * C1 + i) * 9 + uv];
    }
}

// ---------------------------------------------------------------------------
// Kernel 5: 9-tap implicit-GEMM conv on WMMA (bf16 in, f32 accumulate).
// Block tile M=64 x N=256 px (4 rows); wave tile 32x64 (2x4 C frags).
// B tile staged with GLOBAL_LOAD_ASYNC_TO_LDS_B128 (ASYNCcnt), A frags
// sp-scaled into LDS. K loop: 8 chunks of 32 ch x 9 taps.
// ---------------------------------------------------------------------------
__global__ void __launch_bounds__(256) gemm_kernel(
        const __bf16* __restrict__ xs, const __bf16* __restrict__ apack,
        const float* __restrict__ alpha, const float* __restrict__ beta,
        const float* __restrict__ spv, float* __restrict__ out) {
    __shared__ __bf16 ldsB[6 * HP * 32];          // 25,344 B  [row][px][32ch]
    __shared__ __bf16 ldsA[9 * 4 * 32 * 16];      // 36,864 B  sp-scaled A frags

    const int b    = blockIdx.z;
    const int mblk = blockIdx.y;                  // o0 = mblk*64
    const int y0   = blockIdx.x * 4;              // 4 output rows per block
    const int tid  = threadIdx.x;
    const int lane = tid & 31, wid = tid >> 5;
    const int mwave = wid >> 2;                   // 0..1
    const int r     = wid & 3;                    // output row within tile
    const int nlane = lane & 15, khalf = lane >> 4;

    v8f acc[2][4] = {};

    const __bf16* xb = xs + (size_t)b * HP * HP * C1;

    for (int kc = 0; kc < 8; kc++) {
        __syncthreads();
        // ---- stage B: async global->LDS, 16B per lane per issue ----
        for (int idx = tid; idx < 6 * HP * 4; idx += 256) {
            const int q   = idx & 3;              // 8-channel quarter of 32
            const int px  = (idx >> 2) % HP;
            const int row = idx / (HP * 4);
            const __bf16* src = xb +
                ((size_t)(y0 + row) * HP + px) * C1 + kc * 32 + q * 8;
            const unsigned dst =
                (unsigned)(uintptr_t)&ldsB[(row * HP + px) * 32 + q * 8];
            asm volatile("global_load_async_to_lds_b128 %0, %1, off"
                         :: "v"(dst), "v"(src) : "memory");
        }
        // ---- stage A: sp[b,uv]-scaled fragments for this block's 4 M-frags ----
        for (int g = tid; g < 9 * 128; g += 256) {
            const int uv  = g >> 7;
            const int mfl = (g >> 5) & 3;
            const int ln  = g & 31;
            const float sv = spv[b * 9 + uv];
            const v16bf* ap = (const v16bf*)(apack +
                ((((size_t)uv * 16 + (mblk * 4 + mfl)) * 8 + kc) * 32 + ln) * 16);
            v16bf a = *ap;
            #pragma unroll
            for (int e = 0; e < 16; e++) a[e] = (__bf16)((float)a[e] * sv);
            *(v16bf*)&ldsA[(size_t)g * 16] = a;
        }
        asm volatile("s_wait_asynccnt 0x0" ::: "memory");
        __syncthreads();

        // ---- 9 taps x (2 M-frags x 4 N-frags) WMMA ----
        #pragma unroll
        for (int uv = 0; uv < 9; uv++) {
            const int u = uv / 3, v = uv % 3;
            v16bf afr[2];
            #pragma unroll
            for (int mi = 0; mi < 2; mi++)
                afr[mi] = *(const v16bf*)
                    &ldsA[(((uv * 4) + (mwave * 2 + mi)) * 32 + lane) * 16];
            v16bf bfr[4];
            #pragma unroll
            for (int f = 0; f < 4; f++) {
                const int px   = f * 16 + nlane + v;
                const int base = ((r + u) * HP + px) * 32 + khalf * 8;
                const v8bf lo = *(const v8bf*)&ldsB[base];
                const v8bf hi = *(const v8bf*)&ldsB[base + 16];
                #pragma unroll
                for (int e = 0; e < 8; e++) { bfr[f][e] = lo[e]; bfr[f][e + 8] = hi[e]; }
            }
            #pragma unroll
            for (int f = 0; f < 4; f++)
                #pragma unroll
                for (int mi = 0; mi < 2; mi++)
                    acc[mi][f] = __builtin_amdgcn_wmma_f32_16x16x32_bf16(
                        false, afr[mi], false, bfr[f], (short)0, acc[mi][f],
                        false, false);
        }
    }

    // ---- epilogue: z = G*alpha + beta ; out = z*sigmoid(z) (SiLU) ----
    const int y = y0 + r;
    #pragma unroll
    for (int mi = 0; mi < 2; mi++) {
        #pragma unroll
        for (int rr = 0; rr < 8; rr++) {
            const int o = mblk * 64 + mwave * 32 + mi * 16 + rr + khalf * 8;
            const float al = alpha[b * C2 + o];
            const float be = beta[o];
            #pragma unroll
            for (int f = 0; f < 4; f++) {
                const float z = acc[mi][f][rr] * al + be;
                out[(((size_t)b * C2 + o) * H + y) * W + f * 16 + nlane] =
                    z * (1.f / (1.f + __expf(-z)));
            }
        }
    }
}

// ---------------------------------------------------------------------------
extern "C" void kernel_launch(void* const* d_in, const int* in_sizes, int n_in,
                              void* d_out, int out_size, void* d_ws, size_t ws_size,
                              hipStream_t stream) {
    const float* x     = (const float*)d_in[0];
    const float* fc_w  = (const float*)d_in[1];
    const float* g1    = (const float*)d_in[2];
    const float* b1    = (const float*)d_in[3];
    const float* m1    = (const float*)d_in[4];
    const float* v1    = (const float*)d_in[5];
    const float* ch_w  = (const float*)d_in[6];
    const float* ch_b  = (const float*)d_in[7];
    const float* fil_w = (const float*)d_in[8];
    const float* fil_b = (const float*)d_in[9];
    const float* sp_w  = (const float*)d_in[10];
    const float* sp_b  = (const float*)d_in[11];
    // d_in[12]/[13] = ker_w/ker_b: softmax over KN==1 is identically 1.0
    const float* weight = (const float*)d_in[14];
    const float* bias   = (const float*)d_in[15];
    const float* g2     = (const float*)d_in[16];
    const float* b2     = (const float*)d_in[17];
    const float* m2     = (const float*)d_in[18];
    const float* v2     = (const float*)d_in[19];

    char* ws = (char*)d_ws;
    float*  pooled = (float*)(ws + 0);              //  16 KB
    float*  chattn = (float*)(ws + 16384);          //  16 KB
    float*  alpha  = (float*)(ws + 32768);          //  16 KB
    float*  beta   = (float*)(ws + 49152);          //   1 KB
    float*  spv    = (float*)(ws + 50176);          // 576 B (pad to 51200)
    __bf16* apack  = (__bf16*)(ws + 51200);         // 1,179,648 B
    __bf16* xsbuf  = (__bf16*)(ws + 1230848);       // 16*66*66*256*2 = 35,684,352 B

    pool_kernel<<<B * C1, 256, 0, stream>>>(x, pooled);
    attn_kernel<<<1, 256, 0, stream>>>(pooled, fc_w, g1, b1, m1, v1,
                                       ch_w, ch_b, fil_w, fil_b, sp_w, sp_b,
                                       bias, g2, b2, m2, v2,
                                       chattn, alpha, beta, spv);
    packx_kernel<<<B * HP, 256, 0, stream>>>(x, chattn, xsbuf);
    packw_kernel<<<9 * 16 * 8, 32, 0, stream>>>(weight, apack);
    gemm_kernel<<<dim3(16, 4, B), 256, 0, stream>>>(xsbuf, apack, alpha, beta,
                                                    spv, (float*)d_out);
}
